// LSTMClassifier_48945447305756
// MI455X (gfx1250) — matrix operand order repaired and checked by reading, not compiled, not run on hardware
//
#include <hip/hip_runtime.h>
#include <hip/hip_bf16.h>

// ---------------------------------------------------------------------------
// LSTMClassifier for MI455X (gfx1250): mixed-precision WMMA implementation.
//   - f16 operands, f32 accumulate (v_wmma_f32_16x16x32_f16)
//   - weights pre-converted into WMMA B-fragment layout (one-time prep)
//   - per-timestep: lstm0(+head of t-1) -> lstm1(+LayerNorm) ; 2 launches/step
//   - async global->LDS copies (ASYNCcnt) for the f16 hidden-state staging
//   - double-buffered f16 hidden state to avoid cross-block races
// ---------------------------------------------------------------------------

typedef __attribute__((ext_vector_type(16))) _Float16 v16h;
typedef __attribute__((ext_vector_type(8)))  float    v8f;

union Frag16 { v16h v; uint4 q[2]; };

#define UNITS 512
#define BATCH 128
#define TSTEP 1000
#define FEAT  72
#define KPAD0 608   // 96 (x padded) + 512 (h0)
#define KT0   19    // K-tiles of 32
#define KPAD1 1024  // 512 (h0n) + 512 (h1)
#define KT1   32
#define ST4_0 77    // uint4 stride per A-row (odd -> conflict-free ds_load_b128)
#define ST4_1 129

// workspace layout (bytes)
#define OFF_WU0   0ull                          // 19*128 tiles * 1KB = 2,490,368
#define OFF_WU1   2490368ull                    // 32*128 tiles * 1KB = 4,194,304
#define OFF_H0    6684672ull                    // 128*512 f32
#define OFF_C0    6946816ull
#define OFF_H1    7208960ull
#define OFF_C1    7471104ull
#define OFF_H0HA  7733248ull                    // 128*512 f16 (double buffered)
#define OFF_H0HB  7864320ull
#define OFF_H1HA  7995392ull
#define OFF_H1HB  8126464ull
#define OFF_END   8257536ull
#define ZERO_WORDS ((OFF_END - OFF_H0) / 4)     // 393,216

__device__ __forceinline__ float sigm(float x) { return 1.f / (1.f + __expf(-x)); }

// Issue one 16B global -> LDS asynchronous copy (tracked by ASYNCcnt).
__device__ __forceinline__ void async_g2l_b128(const void* gsrc, void* lds_dst) {
  unsigned ldsa = (unsigned)(uintptr_t)lds_dst;   // low 32 bits = LDS byte offset
  asm volatile("global_load_async_to_lds_b128 %0, %1, off"
               :: "v"(ldsa), "v"(gsrc) : "memory");
}
__device__ __forceinline__ void wait_async0() {
  asm volatile("s_wait_asynccnt 0x0" ::: "memory");
}

// ---- prep: f32 weights -> f16 WMMA B-fragment layout ----------------------
// B tile (32K x 16N): element (k,n) -> lane = (k>=16)*16 + n, halfpos = k&15.
// Tile storage: lane-major, 16 halves per lane, 1KB per tile,
// tile index = ct*KT + kt  (ct = column tile 0..127, kt = K tile).

__global__ __launch_bounds__(256) void prep_w0(const float* __restrict__ w0,
                                               const float* __restrict__ u0,
                                               _Float16* __restrict__ dst) {
  int idx = blockIdx.x * 256 + threadIdx.x;
  if (idx >= KPAD0 * 2048) return;
  int k = idx >> 11, c = idx & 2047;
  float v = 0.f;
  if (k < FEAT)      v = w0[k * 2048 + c];
  else if (k >= 96)  v = u0[(k - 96) * 2048 + c];
  int kt = k >> 5, kk = k & 31, ct = c >> 4, n = c & 15;
  int lane = ((kk >> 4) << 4) | n, pos = kk & 15;
  dst[(size_t)(ct * KT0 + kt) * 512 + lane * 16 + pos] = (_Float16)v;
}

__global__ __launch_bounds__(256) void prep_w1(const float* __restrict__ w1,
                                               const float* __restrict__ u1,
                                               _Float16* __restrict__ dst) {
  int idx = blockIdx.x * 256 + threadIdx.x;
  if (idx >= KPAD1 * 2048) return;
  int k = idx >> 11, c = idx & 2047;
  float v = (k < UNITS) ? w1[k * 2048 + c] : u1[(k - UNITS) * 2048 + c];
  int kt = k >> 5, kk = k & 31, ct = c >> 4, n = c & 15;
  int lane = ((kk >> 4) << 4) | n, pos = kk & 15;
  dst[(size_t)(ct * KT1 + kt) * 512 + lane * 16 + pos] = (_Float16)v;
}

__global__ __launch_bounds__(256) void prep_zero(unsigned* __restrict__ p, int n) {
  int i = blockIdx.x * 256 + threadIdx.x;
  if (i < n) p[i] = 0u;
}

// ---- layer 0 LSTM step + classifier head for timestep t-1 -----------------
// grid 32: blockIdx = m*4 + nq ; block = 288 threads = 9 waves.
// Waves 0-7: GEMM (wave w owns unit-tile ut = nq*8+w, gate tiles i,f,g,o).
// Wave 8  : softmax(h1[t-1] @ wf + bf) for 4 batch rows (blockIdx*4..+3).
__global__ __launch_bounds__(288) void lstm0_step(
    const float* __restrict__ x, const float* __restrict__ b0,
    const _Float16* __restrict__ WU0,
    float* __restrict__ h0, float* __restrict__ c0,
    const _Float16* __restrict__ h0h_rd, _Float16* __restrict__ h0h_wr,
    const float* __restrict__ h1prev, const float* __restrict__ wf,
    const float* __restrict__ bfv, float* __restrict__ out, int t) {
  __shared__ uint4 lsA[16 * ST4_0];
  __shared__ float hred[4 * 14];
  const int tid = threadIdx.x;
  const int m = blockIdx.x >> 2;
  const int nq = blockIdx.x & 3;
  _Float16* As = (_Float16*)lsA;
  const int STH = ST4_0 * 8;  // halves per row (616)

  // (a) async LDS fill of the h0 (f16) part: K 96..607, 16B chunks.
  if (tid < 256) {
#pragma unroll
    for (int it = 0; it < 4; ++it) {
      int chunk = tid + 256 * it;            // 0..1023
      int row = chunk >> 6, j = chunk & 63;  // j: 8-half group
      int b = m * 16 + row;
      async_g2l_b128(h0h_rd + ((size_t)b * UNITS + j * 8),
                     As + row * STH + 96 + j * 8);
    }
  }
  // (b) compute staging of x_t (f32->f16, pad 72->96): K 0..95.
  for (int i = tid; i < 16 * 96; i += 288) {
    int row = i / 96, k = i - row * 96;
    int b = m * 16 + row;
    As[row * STH + k] =
        (_Float16)((k < FEAT) ? x[(size_t)(b * TSTEP + t) * FEAT + k] : 0.f);
  }
  wait_async0();
  __syncthreads();

  const int wave = tid >> 5, lane = tid & 31;
  if (wave < 8) {
    const int ut = nq * 8 + wave;
    const int half = lane >> 4, nn = lane & 15, arow = lane & 15;
    const v8f vz = {0.f, 0.f, 0.f, 0.f, 0.f, 0.f, 0.f, 0.f};
    v8f acc[4] = {vz, vz, vz, vz};

    for (int kt = 0; kt < KT0; ++kt) {
      Frag16 a;
      int ai = arow * ST4_0 + kt * 4 + half * 2;
      a.q[0] = lsA[ai];
      a.q[1] = lsA[ai + 1];
#pragma unroll
      for (int g = 0; g < 4; ++g) {
        const uint4* bp = (const uint4*)(WU0 + (size_t)((g * 32 + ut) * KT0 + kt) * 512);
        if (kt + 1 < KT0)
          __builtin_prefetch(WU0 + (size_t)((g * 32 + ut) * KT0 + kt + 1) * 512, 0, 2);
        Frag16 bf;
        bf.q[0] = bp[lane * 2];
        bf.q[1] = bp[lane * 2 + 1];
        acc[g] = __builtin_amdgcn_wmma_f32_16x16x32_f16(false, a.v, false, bf.v,
                                                        (short)0, acc[g], false, false);
      }
    }

    // C layout: vgpr r, lanes 0-15 -> M=r, N=lane ; lanes 16-31 -> M=8+r.
#pragma unroll
    for (int r = 0; r < 8; ++r) {
      int row = r + half * 8;
      int b = m * 16 + row;
      int unit = ut * 16 + nn;
      size_t o = (size_t)b * UNITS + unit;
      float iv = sigm(acc[0][r] + b0[unit]);
      float fv = sigm(acc[1][r] + b0[UNITS + unit]);
      float gv = tanhf(acc[2][r] + b0[2 * UNITS + unit]);
      float ov = sigm(acc[3][r] + b0[3 * UNITS + unit]);
      float cv = fv * c0[o] + iv * gv;
      float hv = ov * tanhf(cv);
      c0[o] = cv;
      h0[o] = hv;
      h0h_wr[o] = (_Float16)hv;
    }
  } else if (t > 0) {
    // Classifier head for timestep t-1 (h1prev fully written by prior kernel).
    for (int p = lane; p < 56; p += 32) {
      int row = p / 14, c = p - row * 14;
      int b = blockIdx.x * 4 + row;
      float acc = bfv[c];
      for (int k = 0; k < UNITS; ++k)
        acc += h1prev[(size_t)b * UNITS + k] * wf[k * 14 + c];
      hred[row * 14 + c] = acc;
    }
    asm volatile("s_wait_dscnt 0x0" ::: "memory");
    __builtin_amdgcn_wave_barrier();
    if (lane < 4) {
      int b = blockIdx.x * 4 + lane;
      float l[14], mx = -1e30f;
#pragma unroll
      for (int c = 0; c < 14; ++c) { l[c] = hred[lane * 14 + c]; mx = fmaxf(mx, l[c]); }
      float s = 0.f;
#pragma unroll
      for (int c = 0; c < 14; ++c) { l[c] = __expf(l[c] - mx); s += l[c]; }
      float inv = 1.f / s;
      float* o = out + (size_t)(b * TSTEP + (t - 1)) * 14;
#pragma unroll
      for (int c = 0; c < 14; ++c) o[c] = l[c] * inv;
    }
  }
}

// ---- layer 1 LSTM step (with fused LayerNorm on h0) -----------------------
__global__ __launch_bounds__(256) void lstm1_step(
    const float* __restrict__ b1, const float* __restrict__ gamma,
    const float* __restrict__ beta, const _Float16* __restrict__ WU1,
    const float* __restrict__ h0,
    float* __restrict__ h1, float* __restrict__ c1,
    const _Float16* __restrict__ h1h_rd, _Float16* __restrict__ h1h_wr) {
  __shared__ uint4 lsA[16 * ST4_1];
  __shared__ float mrs[16 * 2];  // mu, rsig per row
  const int tid = threadIdx.x;
  const int m = blockIdx.x >> 2;
  const int nq = blockIdx.x & 3;
  _Float16* As = (_Float16*)lsA;
  const int STH = ST4_1 * 8;  // halves per row (1032)

  // (a) async LDS fill of the h1 (f16) part: K 512..1023.
#pragma unroll
  for (int it = 0; it < 4; ++it) {
    int chunk = tid + 256 * it;            // 0..1023
    int row = chunk >> 6, j = chunk & 63;
    int b = m * 16 + row;
    async_g2l_b128(h1h_rd + ((size_t)b * UNITS + j * 8),
                   As + row * STH + UNITS + j * 8);
  }

  // (b) LayerNorm stats over the fully-written f32 h0 (prior kernel).
  if (tid < 16) {
    int b = m * 16 + tid;
    float s = 0.f, ss = 0.f;
    for (int k = 0; k < UNITS; ++k) {
      float v = h0[(size_t)b * UNITS + k];
      s += v;
      ss += v * v;
    }
    float mu = s * (1.f / UNITS);
    float var = ss * (1.f / UNITS) - mu * mu;
    mrs[tid * 2] = mu;
    mrs[tid * 2 + 1] = rsqrtf(var + 1e-3f);
  }
  __syncthreads();

  // (c) compute staging of LayerNorm(h0)*gamma+beta: K 0..511.
  for (int i = tid; i < 16 * UNITS; i += 256) {
    int row = i >> 9, k = i & 511;
    int b = m * 16 + row;
    float hv = h0[(size_t)b * UNITS + k];
    As[row * STH + k] =
        (_Float16)((hv - mrs[row * 2]) * mrs[row * 2 + 1] * gamma[k] + beta[k]);
  }
  wait_async0();
  __syncthreads();

  const int wave = tid >> 5, lane = tid & 31;
  const int ut = nq * 8 + wave;
  const int half = lane >> 4, nn = lane & 15, arow = lane & 15;
  const v8f vz = {0.f, 0.f, 0.f, 0.f, 0.f, 0.f, 0.f, 0.f};
  v8f acc[4] = {vz, vz, vz, vz};

  for (int kt = 0; kt < KT1; ++kt) {
    Frag16 a;
    int ai = arow * ST4_1 + kt * 4 + half * 2;
    a.q[0] = lsA[ai];
    a.q[1] = lsA[ai + 1];
#pragma unroll
    for (int g = 0; g < 4; ++g) {
      const uint4* bp = (const uint4*)(WU1 + (size_t)((g * 32 + ut) * KT1 + kt) * 512);
      if (kt + 1 < KT1)
        __builtin_prefetch(WU1 + (size_t)((g * 32 + ut) * KT1 + kt + 1) * 512, 0, 2);
      Frag16 bf;
      bf.q[0] = bp[lane * 2];
      bf.q[1] = bp[lane * 2 + 1];
      acc[g] = __builtin_amdgcn_wmma_f32_16x16x32_f16(false, a.v, false, bf.v,
                                                      (short)0, acc[g], false, false);
    }
  }

#pragma unroll
  for (int r = 0; r < 8; ++r) {
    int row = r + half * 8;
    int b = m * 16 + row;
    int unit = ut * 16 + nn;
    size_t o = (size_t)b * UNITS + unit;
    float iv = sigm(acc[0][r] + b1[unit]);
    float fv = sigm(acc[1][r] + b1[UNITS + unit]);
    float gv = tanhf(acc[2][r] + b1[2 * UNITS + unit]);
    float ov = sigm(acc[3][r] + b1[3 * UNITS + unit]);
    float cv = fv * c1[o] + iv * gv;
    float hv = ov * tanhf(cv);
    c1[o] = cv;
    h1[o] = hv;
    h1h_wr[o] = (_Float16)hv;
  }
}

// ---- standalone classifier head (used once, for the final timestep) -------
__global__ __launch_bounds__(64) void head_step(
    const float* __restrict__ h1, const float* __restrict__ wf,
    const float* __restrict__ bf, float* __restrict__ out, int t) {
  __shared__ float red[64 * 14];
  const int b = blockIdx.x, tid = threadIdx.x;
  float acc[14];
#pragma unroll
  for (int c = 0; c < 14; ++c) acc[c] = 0.f;
  for (int k = tid; k < UNITS; k += 64) {
    float h = h1[(size_t)b * UNITS + k];
#pragma unroll
    for (int c = 0; c < 14; ++c) acc[c] += h * wf[k * 14 + c];
  }
#pragma unroll
  for (int c = 0; c < 14; ++c) red[tid * 14 + c] = acc[c];
  __syncthreads();
  for (int off = 32; off > 0; off >>= 1) {
    if (tid < off)
      for (int c = 0; c < 14; ++c) red[tid * 14 + c] += red[(tid + off) * 14 + c];
    __syncthreads();
  }
  if (tid == 0) {
    float l[14], mx = -1e30f;
#pragma unroll
    for (int c = 0; c < 14; ++c) { l[c] = red[c] + bf[c]; mx = fmaxf(mx, l[c]); }
    float s = 0.f;
#pragma unroll
    for (int c = 0; c < 14; ++c) { l[c] = __expf(l[c] - mx); s += l[c]; }
    float inv = 1.f / s;
    float* o = out + (size_t)(b * TSTEP + t) * 14;
#pragma unroll
    for (int c = 0; c < 14; ++c) o[c] = l[c] * inv;
  }
}

// ---------------------------------------------------------------------------
extern "C" void kernel_launch(void* const* d_in, const int* in_sizes, int n_in,
                              void* d_out, int out_size, void* d_ws, size_t ws_size,
                              hipStream_t stream) {
  const float* x     = (const float*)d_in[0];
  const float* w0    = (const float*)d_in[1];
  const float* u0    = (const float*)d_in[2];
  const float* b0    = (const float*)d_in[3];
  const float* gamma = (const float*)d_in[4];
  const float* beta  = (const float*)d_in[5];
  const float* w1    = (const float*)d_in[6];
  const float* u1    = (const float*)d_in[7];
  const float* b1    = (const float*)d_in[8];
  const float* wf    = (const float*)d_in[9];
  const float* bfv   = (const float*)d_in[10];

  char* ws = (char*)d_ws;
  _Float16* WU0 = (_Float16*)(ws + OFF_WU0);
  _Float16* WU1 = (_Float16*)(ws + OFF_WU1);
  float* h0 = (float*)(ws + OFF_H0);
  float* c0 = (float*)(ws + OFF_C0);
  float* h1 = (float*)(ws + OFF_H1);
  float* c1 = (float*)(ws + OFF_C1);
  _Float16* h0h[2] = {(_Float16*)(ws + OFF_H0HA), (_Float16*)(ws + OFF_H0HB)};
  _Float16* h1h[2] = {(_Float16*)(ws + OFF_H1HA), (_Float16*)(ws + OFF_H1HB)};
  float* out = (float*)d_out;

  prep_w0<<<(KPAD0 * 2048 + 255) / 256, 256, 0, stream>>>(w0, u0, WU0);
  prep_w1<<<(KPAD1 * 2048 + 255) / 256, 256, 0, stream>>>(w1, u1, WU1);
  prep_zero<<<(ZERO_WORDS + 255) / 256, 256, 0, stream>>>((unsigned*)(ws + OFF_H0),
                                                          (int)ZERO_WORDS);

  for (int t = 0; t < TSTEP; ++t) {
    const int rb = t & 1, wb = rb ^ 1;
    lstm0_step<<<32, 288, 0, stream>>>(x, b0, WU0, h0, c0, h0h[rb], h0h[wb],
                                       h1, wf, bfv, out, t);
    lstm1_step<<<32, 256, 0, stream>>>(b1, gamma, beta, WU1, h0, h1, c1,
                                       h1h[rb], h1h[wb]);
  }
  head_step<<<128, 64, 0, stream>>>(h1, wf, bfv, out, TSTEP - 1);
}
